// AtomisticGraphGNN_79534204388000
// MI455X (gfx1250) — matrix operand order repaired and checked by reading, not compiled, not run on hardware
//
#include <hip/hip_runtime.h>

// GAT (3 layers, H=4, C=64) for MI455X / gfx1250.
// GEMMs: v_wmma_f32_16x16x32_f16, fully unrolled K, vectorized fragment loads,
// B pre-packed into lane-major fragment layout.
// Edge softmax + aggregation: memory/atomic-bound streaming kernels.

typedef __attribute__((ext_vector_type(16))) _Float16 v16h;
typedef __attribute__((ext_vector_type(8)))  _Float16 v8h;
typedef __attribute__((ext_vector_type(8)))  float    v8f;

#define HEADS 4
#define CH    64
#define HC    256      // HEADS*CH
#define DIN   32
#define NLAYERS 3
#define NGR   256

// ---------------- utility kernels ----------------

__global__ void k_f32_to_f16(const float* __restrict__ src, _Float16* __restrict__ dst, int n) {
  int i = blockIdx.x * blockDim.x + threadIdx.x;
  if (i < n) dst[i] = (_Float16)src[i];
}

// Pack B[K,Ncols] (f32, row-major) into per-wave fragment order (f16):
//   fragment element e of lane L in k-step ks maps to  k = ks*32 + (L>>4)*16 + e,
//   col = colTile*16 + (L&15)  (contiguous k order, same as the A fragment loads).
// Packed index: (((colTile*(K/32) + ks)*32 + lane)*16 + e  -> each lane's fragment
// is 16 contiguous halfs = one 32-byte load in the GEMM.
__global__ void k_pack_b(const float* __restrict__ B, _Float16* __restrict__ Bp,
                         int K, int Ncols) {
  int t = blockIdx.x * blockDim.x + threadIdx.x;
  if (t >= K * Ncols) return;
  int k = t / Ncols, col = t - k * Ncols;
  int ks = k >> 5, rem = k & 31;
  int hi = rem >> 4, e = rem & 15;
  int lane = (hi << 4) | (col & 15);
  int colTile = col >> 4;
  size_t p = (((size_t)colTile * (K >> 5) + ks) * 32 + lane) * 16 + e;
  Bp[p] = (_Float16)B[t];
}

// ---------------- WMMA GEMM: C[M,Ncols] = act(A[M,K] @ Bpacked + bias) ----------------
// One wave computes a 16x16 tile; K/BIAS/RELU compile-time -> fully unrolled.
// A 16x32 f16 fragment: row = lane&15; per K-step the lane's halfs are one
// contiguous run of 16 (k0+hi*16 .. +15) -> two b128 loads.
template <int K, bool BIAS, bool RELU>
__global__ void k_wmma_gemm(const _Float16* __restrict__ A, const _Float16* __restrict__ Bp,
                            float* __restrict__ Cout, const float* __restrict__ bias,
                            int M, int Ncols) {
  constexpr int KSTEPS = K / 32;
  const int lane = threadIdx.x;
  const int r    = lane & 15;
  const int hi   = lane >> 4;
  const int row0 = blockIdx.x * 16;
  const int col0 = blockIdx.y * 16;

  const int rowA = (row0 + r < M) ? (row0 + r) : (M - 1);
  const v8h*  arow = (const v8h*)(A + (size_t)rowA * K);
  const v16h* bfrag = (const v16h*)Bp + ((size_t)blockIdx.y * KSTEPS) * 32 + lane;

  v8f acc = {};
#pragma unroll
  for (int ks = 0; ks < KSTEPS; ++ks) {
    union { v16h v; v8h h[2]; } ua;
    ua.h[0] = arow[ks * 4 + hi * 2];        // K = ks*32 + hi*16 .. +7
    ua.h[1] = arow[ks * 4 + hi * 2 + 1];    // K = ks*32 + hi*16 + 8 .. +15
    v16h b = bfrag[ks * 32];
    acc = __builtin_amdgcn_wmma_f32_16x16x32_f16(false, ua.v, false, b, (short)0, acc,
                                                 false, false);
  }

  const int col = col0 + r;
  float bv = 0.0f;
  if constexpr (BIAS) bv = bias[col];
  if (row0 + 16 <= M) {
#pragma unroll
    for (int v = 0; v < 8; ++v) {
      float val = acc[v] + bv;
      if constexpr (RELU) val = fmaxf(val, 0.0f);
      Cout[(size_t)(row0 + v + hi * 8) * Ncols + col] = val;
    }
  } else {
#pragma unroll
    for (int v = 0; v < 8; ++v) {
      int row = row0 + v + hi * 8;
      if (row < M) {
        float val = acc[v] + bv;
        if constexpr (RELU) val = fmaxf(val, 0.0f);
        Cout[(size_t)row * Ncols + col] = val;
      }
    }
  }
}

// ---------------- attention scalar projections: al_s / al_d [N,H] ----------------
__global__ void k_alpha(const float* __restrict__ xh, const float* __restrict__ asrc,
                        const float* __restrict__ adst, float* __restrict__ als,
                        float* __restrict__ ald, int N) {
  int t = blockIdx.x * blockDim.x + threadIdx.x;
  if (t >= N * HEADS) return;
  int n = t >> 2, h = t & 3;
  const float* xr = xh + (size_t)n * HC + h * CH;
  const float* as = asrc + h * CH;
  const float* ad = adst + h * CH;
  float s = 0.f, d = 0.f;
#pragma unroll 8
  for (int c = 0; c < CH; ++c) { float v = xr[c]; s += v * as[c]; d += v * ad[c]; }
  als[t] = s; ald[t] = d;
}

__global__ void k_init_layer(float* __restrict__ accum, float* __restrict__ emax,
                             float* __restrict__ den, int N) {
  int i = blockIdx.x * blockDim.x + threadIdx.x;
  if (i < N * HC) accum[i] = 0.f;
  if (i < N * HEADS) { emax[i] = -__builtin_inff(); den[i] = 0.f; }
}

// sign-aware float atomic max via integer atomics (correct for init = -inf)
__device__ __forceinline__ void atomicMaxF(float* addr, float value) {
  if (value >= 0.f) atomicMax((int*)addr, __float_as_int(value));
  else              atomicMin((unsigned int*)addr, __float_as_uint(value));
}

__device__ __forceinline__ void edge_sd(const int* __restrict__ ei, int E, int e, int& s, int& d) {
  if (e < E) { s = ei[e]; d = ei[E + e]; }   // edge_index rows [2,E], flat
  else       { s = e - E; d = e - E; }        // appended self-loop
}

__global__ void k_edge_max(const int* __restrict__ ei, int E, int N,
                           const float* __restrict__ als, const float* __restrict__ ald,
                           float* __restrict__ emax) {
  int t = blockIdx.x * blockDim.x + threadIdx.x;
  int ET = E + N;
  if (t >= ET * HEADS) return;
  int e = t >> 2, h = t & 3, s, d;
  edge_sd(ei, E, e, s, d);
  float v = als[s * HEADS + h] + ald[d * HEADS + h];
  v = v > 0.f ? v : 0.2f * v;                 // leaky_relu(0.2)
  atomicMaxF(&emax[d * HEADS + h], v);
}

__global__ void k_edge_exp(const int* __restrict__ ei, int E, int N,
                           const float* __restrict__ als, const float* __restrict__ ald,
                           const float* __restrict__ emax, float* __restrict__ den,
                           float* __restrict__ exbuf) {
  int t = blockIdx.x * blockDim.x + threadIdx.x;
  int ET = E + N;
  if (t >= ET * HEADS) return;
  int e = t >> 2, h = t & 3, s, d;
  edge_sd(ei, E, e, s, d);
  float v = als[s * HEADS + h] + ald[d * HEADS + h];
  v = v > 0.f ? v : 0.2f * v;
  float ex = __expf(v - emax[d * HEADS + h]);
  exbuf[t] = ex;
  atomicAdd(&den[d * HEADS + h], ex);
}

// per-(edge, channel): 4 weighted atomic adds (one per head)
__global__ void k_edge_msg(const int* __restrict__ ei, int E, int N,
                           const float* __restrict__ xh, const float* __restrict__ exbuf,
                           const float* __restrict__ den, float* __restrict__ accum) {
  long long t = (long long)blockIdx.x * blockDim.x + threadIdx.x;
  long long ET = (long long)(E + N);
  if (t >= ET * CH) return;
  int e = (int)(t >> 6);
  int c = (int)(t & 63);
  int s, d;
  edge_sd(ei, E, e, s, d);
  const float* xr = xh + (size_t)s * HC + c;
  float*       ar = accum + (size_t)d * HC + c;
  __builtin_prefetch(xr, 0, 0);               // global_prefetch_b8
#pragma unroll
  for (int h = 0; h < HEADS; ++h) {
    float w = exbuf[(size_t)e * HEADS + h] / (den[d * HEADS + h] + 1e-16f);
    atomicAdd(ar + h * CH, xr[h * CH] * w);
  }
}

// head-mean + bias -> LayerNorm -> ReLU -> (+residual); one wave32 per node
__global__ void k_finalize(const float* __restrict__ accum, const float* __restrict__ bgat,
                           const float* __restrict__ lng, const float* __restrict__ lnb,
                           float* __restrict__ h, int N, int residual) {
  int warp = threadIdx.x >> 5;
  int lane = threadIdx.x & 31;
  int n = blockIdx.x * (blockDim.x >> 5) + warp;
  if (n >= N) return;
  const float* ac = accum + (size_t)n * HC;
  int c0 = lane, c1 = lane + 32;
  float y0 = 0.f, y1 = 0.f;
#pragma unroll
  for (int hh = 0; hh < HEADS; ++hh) { y0 += ac[hh * CH + c0]; y1 += ac[hh * CH + c1]; }
  y0 = y0 * 0.25f + bgat[c0];
  y1 = y1 * 0.25f + bgat[c1];
  float s = y0 + y1;
#pragma unroll
  for (int off = 16; off > 0; off >>= 1) s += __shfl_xor(s, off, 32);
  float mu = s * (1.0f / 64.0f);
  float d0 = y0 - mu, d1 = y1 - mu;
  float v = d0 * d0 + d1 * d1;
#pragma unroll
  for (int off = 16; off > 0; off >>= 1) v += __shfl_xor(v, off, 32);
  float rstd = rsqrtf(v * (1.0f / 64.0f) + 1e-5f);
  float o0 = fmaxf(d0 * rstd * lng[c0] + lnb[c0], 0.f);
  float o1 = fmaxf(d1 * rstd * lng[c1] + lnb[c1], 0.f);
  float* hr = h + (size_t)n * CH;
  if (residual) { o0 += hr[c0]; o1 += hr[c1]; }
  hr[c0] = o0; hr[c1] = o1;
}

// ---------------- pooling + output head ----------------
__global__ void k_pool_init(float* __restrict__ sums, float* __restrict__ cnt) {
  int i = blockIdx.x * blockDim.x + threadIdx.x;
  if (i < NGR * CH) sums[i] = 0.f;
  if (i < NGR) cnt[i] = 0.f;
}

__global__ void k_pool(const float* __restrict__ h, const int* __restrict__ batch,
                       float* __restrict__ sums, float* __restrict__ cnt, int N) {
  int t = blockIdx.x * blockDim.x + threadIdx.x;
  if (t >= N * CH) return;
  int n = t >> 6, c = t & 63;
  int g = batch[n];
  atomicAdd(&sums[g * CH + c], h[(size_t)n * CH + c]);
  if (c == 0) atomicAdd(&cnt[g], 1.f);
}

__global__ void k_output(const float* __restrict__ sums, const float* __restrict__ cnt,
                         const float* __restrict__ Wout, const float* __restrict__ bout,
                         float* __restrict__ out) {
  int t = blockIdx.x * blockDim.x + threadIdx.x;
  if (t >= NGR * 32) return;
  int g = t >> 5, o = t & 31;
  float inv = 1.f / fmaxf(cnt[g], 1.f);
  float acc = bout[o];
#pragma unroll 8
  for (int c = 0; c < CH; ++c) acc += sums[g * CH + c] * inv * Wout[c * 32 + o];
  out[t] = acc;
}

// ---------------- driver ----------------
extern "C" void kernel_launch(void* const* d_in, const int* in_sizes, int n_in,
                              void* d_out, int out_size, void* d_ws, size_t ws_size,
                              hipStream_t stream) {
  const float* x       = (const float*)d_in[0];
  const int*   ei      = (const int*)d_in[1];
  const int*   batch   = (const int*)d_in[2];
  const float* W_in    = (const float*)d_in[3];
  const float* b_in    = (const float*)d_in[4];
  const float* W_gat   = (const float*)d_in[5];
  const float* att_src = (const float*)d_in[6];
  const float* att_dst = (const float*)d_in[7];
  const float* b_gat   = (const float*)d_in[8];
  const float* ln_g    = (const float*)d_in[9];
  const float* ln_b    = (const float*)d_in[10];
  const float* W_out   = (const float*)d_in[11];
  const float* b_out   = (const float*)d_in[12];
  float* out = (float*)d_out;

  const int N  = in_sizes[0] / DIN;
  const int E  = in_sizes[1] / 2;
  const int ET = E + N;

  // workspace carve-out
  char* p = (char*)d_ws;
  auto alloc = [&](size_t bytes) -> void* {
    void* r = (void*)p;
    p += (bytes + 255) & ~(size_t)255;
    return r;
  };
  float*    h      = (float*)alloc((size_t)N * CH * 4);
  float*    xh     = (float*)alloc((size_t)N * HC * 4);
  float*    accum  = (float*)alloc((size_t)N * HC * 4);
  _Float16* hf16   = (_Float16*)alloc((size_t)N * CH * 2);
  _Float16* xf16   = (_Float16*)alloc((size_t)N * DIN * 2);
  _Float16* wiPk   = (_Float16*)alloc((size_t)DIN * CH * 2);              // packed W_in
  _Float16* wgPk   = (_Float16*)alloc((size_t)NLAYERS * CH * HC * 2);     // packed W_gat
  float*    als    = (float*)alloc((size_t)N * HEADS * 4);
  float*    ald    = (float*)alloc((size_t)N * HEADS * 4);
  float*    emax   = (float*)alloc((size_t)N * HEADS * 4);
  float*    den    = (float*)alloc((size_t)N * HEADS * 4);
  float*    exbuf  = (float*)alloc((size_t)ET * HEADS * 4);
  float*    sums   = (float*)alloc((size_t)NGR * CH * 4);
  float*    cnt    = (float*)alloc((size_t)NGR * 4);

  const int TB = 256;
  auto cdiv = [](long long a, long long b) { return (int)((a + b - 1) / b); };

  // input conversion + weight packing (f32 -> f16 fragment order)
  k_f32_to_f16<<<cdiv((long long)N * DIN, TB), TB, 0, stream>>>(x, xf16, N * DIN);
  k_pack_b<<<cdiv(DIN * CH, TB), TB, 0, stream>>>(W_in, wiPk, DIN, CH);
  for (int i = 0; i < NLAYERS; ++i)
    k_pack_b<<<cdiv(CH * HC, TB), TB, 0, stream>>>(W_gat + (size_t)i * CH * HC,
                                                   wgPk + (size_t)i * CH * HC, CH, HC);

  // h = relu(x @ W_in + b_in)   [N,32]x[32,64]
  {
    dim3 grid(cdiv(N, 16), CH / 16);
    k_wmma_gemm<DIN, true, true><<<grid, 32, 0, stream>>>(xf16, wiPk, h, b_in, N, CH);
  }

  for (int i = 0; i < NLAYERS; ++i) {
    // xh = h @ W_gat[i]  [N,64]x[64,256]
    k_f32_to_f16<<<cdiv((long long)N * CH, TB), TB, 0, stream>>>(h, hf16, N * CH);
    {
      dim3 grid(cdiv(N, 16), HC / 16);
      k_wmma_gemm<CH, false, false><<<grid, 32, 0, stream>>>(
          hf16, wgPk + (size_t)i * CH * HC, xh, nullptr, N, HC);
    }
    k_alpha<<<cdiv((long long)N * HEADS, TB), TB, 0, stream>>>(
        xh, att_src + (size_t)i * HEADS * CH, att_dst + (size_t)i * HEADS * CH, als, ald, N);
    k_init_layer<<<cdiv((long long)N * HC, TB), TB, 0, stream>>>(accum, emax, den, N);
    k_edge_max<<<cdiv((long long)ET * HEADS, TB), TB, 0, stream>>>(ei, E, N, als, ald, emax);
    k_edge_exp<<<cdiv((long long)ET * HEADS, TB), TB, 0, stream>>>(ei, E, N, als, ald, emax, den, exbuf);
    k_edge_msg<<<cdiv((long long)ET * CH, TB), TB, 0, stream>>>(ei, E, N, xh, exbuf, den, accum);
    k_finalize<<<cdiv(N, TB / 32), TB, 0, stream>>>(
        accum, b_gat + (size_t)i * CH, ln_g + (size_t)i * CH, ln_b + (size_t)i * CH,
        h, N, (i > 0) ? 1 : 0);
  }

  // global mean pool + output projection
  k_pool_init<<<cdiv(NGR * CH, TB), TB, 0, stream>>>(sums, cnt);
  k_pool<<<cdiv((long long)N * CH, TB), TB, 0, stream>>>(h, batch, sums, cnt, N);
  k_output<<<cdiv(NGR * 32, TB), TB, 0, stream>>>(sums, cnt, W_out, b_out, out);
}
